// Block_28329604284812
// MI455X (gfx1250) — matrix operand order repaired and checked by reading, not compiled, not run on hardware
//
#include <hip/hip_runtime.h>
#include <hip/hip_bf16.h>

// ---------------- problem constants ----------------
#define DIMX 2048
#define NH   32
#define DN   128
#define DR   64
#define DV   128
#define KVL  512
#define SEQ  2048
#define NQ   (NH * (DN + DR))        // 6144
#define KVA  (KVL + DR)              // 576
#define ODIM (NH * DV)               // 4096
#define SCALE_F 0.07216878364870323f // (DN+DR)^-0.5 = 1/sqrt(192)

typedef unsigned short u16;
typedef unsigned int   u32;
typedef __attribute__((ext_vector_type(16))) __bf16 v16bf;
typedef __attribute__((ext_vector_type(8)))  float  v8f;
typedef __attribute__((ext_vector_type(4)))  unsigned int v4u;
typedef __attribute__((ext_vector_type(8)))  int v8i;
typedef __attribute__((ext_vector_type(4)))  int v4i;

union FragAB { v16bf v; u32 u[8]; };

__device__ __forceinline__ u16 f2bf(float f) {
  u32 x = __float_as_uint(f);
  u32 r = (x + 0x7FFFu + ((x >> 16) & 1u)) >> 16;   // round-to-nearest-even
  return (u16)r;
}
__device__ __forceinline__ float bf2f(u16 b) { return __uint_as_float(((u32)b) << 16); }

__device__ __forceinline__ v8f wmma_bf16(const FragAB& a, const FragAB& b, v8f c) {
  return __builtin_amdgcn_wmma_f32_16x16x32_bf16(false, a.v, false, b.v,
                                                 (short)0, c, false, false);
}

// ---- Tensor Data Mover: 2D tile (rows x cols bf16) global -> LDS ----
// D# per CDNA5 ISA ch.8: group0 = {count/type/lds_addr/global_addr},
// group1 = {data_size, tensor_dim0/1, tile_dim0/1, tensor_dim0_stride}.
__device__ __forceinline__ void tdm_load_2d(const u16* gsrc, u32 ldsOff,
                                            int rows, int cols,
                                            int strideElems, int tensorRows) {
  unsigned long long ga = (unsigned long long)gsrc;
  v4u g0;
  g0[0] = 1u;                                  // count=1, user descriptor
  g0[1] = ldsOff;                              // lds_addr (bytes)
  g0[2] = (u32)ga;                             // global_addr[31:0]
  g0[3] = (u32)(ga >> 32) | (2u << 30);        // global_addr[56:32] | type=2
  v8i g1;
  g1[0] = 0x00010000;                          // data_size=1 (2 bytes/elem)
  g1[1] = (int)(((u32)strideElems & 0xFFFFu) << 16);          // tensor_dim0 lo
  g1[2] = (int)(((u32)strideElems >> 16) |
                (((u32)tensorRows & 0xFFFFu) << 16));         // td0 hi | td1 lo
  g1[3] = (int)(((u32)tensorRows >> 16) |
                (((u32)cols & 0xFFFFu) << 16));               // td1 hi | tile_dim0
  g1[4] = (int)((u32)rows & 0xFFFFu);                         // tile_dim1 (tile_dim2=0)
  g1[5] = strideElems;                                        // tensor_dim0_stride lo32
  g1[6] = 0;
  g1[7] = 0;
  v4i z4 = {0, 0, 0, 0};
#if __clang_major__ >= 23
  v8i z8 = {0, 0, 0, 0, 0, 0, 0, 0};
  __builtin_amdgcn_tensor_load_to_lds(g0, g1, z4, z4, z8, 0);
#else
  __builtin_amdgcn_tensor_load_to_lds(g0, g1, z4, z4, 0);
#endif
}

// ---------------- prep kernels ----------------
__global__ void k_f32_to_bf16(const float* __restrict__ s, u16* __restrict__ d, long n) {
  long i = blockIdx.x * (long)blockDim.x + threadIdx.x;
  if (i < n) d[i] = f2bf(s[i]);
}

// src [K][N] f32  ->  dst [N][K] bf16  (transpose-convert)
__global__ void k_convert_t(const float* __restrict__ s, u16* __restrict__ d, int K, int N) {
  long i = blockIdx.x * (long)blockDim.x + threadIdx.x;
  long total = (long)K * N;
  if (i >= total) return;
  int  n = (int)(i % N);
  long k = i / N;
  d[(long)n * K + k] = f2bf(s[i]);
}

// wkv_b [NH*(DN+DV)][KVL] -> wbk_t [NH][KVL][DN]  (B^T for q-absorb)
//                            wbv   [NH][DV][KVL]  (already B^T for v-absorb)
__global__ void k_split_wb(const float* __restrict__ wb,
                           u16* __restrict__ wbk_t, u16* __restrict__ wbv) {
  long i = blockIdx.x * (long)blockDim.x + threadIdx.x;
  long total = (long)NH * (DN + DV) * KVL;
  if (i >= total) return;
  int  c   = (int)(i % KVL);
  long rem = i / KVL;
  int  d   = (int)(rem % (DN + DV));
  int  h   = (int)(rem / (DN + DV));
  u16 v = f2bf(wb[i]);
  if (d < DN) wbk_t[((long)h * KVL + c) * DN + d] = v;
  else        wbv[((long)h * DV + (d - DN)) * KVL + c] = v;
}

// q [S][NQ] bf16 -> qn [NH][S][DN], qp (RoPE'd) [NH][S][DR]
__global__ void k_rope_q(const u16* __restrict__ q,
                         const float* __restrict__ fc, const float* __restrict__ fs,
                         u16* __restrict__ qn, u16* __restrict__ qp) {
  int idx = blockIdx.x * blockDim.x + threadIdx.x;     // over S*NH
  if (idx >= SEQ * NH) return;
  int s = idx / NH, h = idx % NH;
  const u16* src = q + (long)s * NQ + h * (DN + DR);
  u16* dn = qn + ((long)h * SEQ + s) * DN;
  for (int d = 0; d < DN; d++) dn[d] = src[d];
  u16* dp = qp + ((long)h * SEQ + s) * DR;
  for (int i = 0; i < DR / 2; i++) {
    float xr = bf2f(src[DN + 2 * i]), xi = bf2f(src[DN + 2 * i + 1]);
    float c = fc[s * (DR / 2) + i], sn = fs[s * (DR / 2) + i];
    dp[2 * i]     = f2bf(xr * c - xi * sn);
    dp[2 * i + 1] = f2bf(xr * sn + xi * c);
  }
}

// kv_full [S][576] f32 -> kvc [S][KVL] bf16 (rms-norm), kpe [S][DR] bf16 (RoPE)
__global__ __launch_bounds__(256)
void k_kv_post(const float* __restrict__ kv, const float* __restrict__ w,
               const float* __restrict__ fc, const float* __restrict__ fs,
               u16* __restrict__ kvc, u16* __restrict__ kpe) {
  __shared__ float red[256];
  int s = blockIdx.x;
  const float* row = kv + (long)s * KVA;
  float ss = 0.f;
  for (int i = threadIdx.x; i < KVL; i += 256) { float v = row[i]; ss += v * v; }
  red[threadIdx.x] = ss;
  __syncthreads();
  for (int st = 128; st > 0; st >>= 1) {
    if ((int)threadIdx.x < st) red[threadIdx.x] += red[threadIdx.x + st];
    __syncthreads();
  }
  float inv = rsqrtf(red[0] / (float)KVL + 1e-6f);
  for (int i = threadIdx.x; i < KVL; i += 256)
    kvc[(long)s * KVL + i] = f2bf(row[i] * inv * w[i]);
  if (threadIdx.x < DR / 2) {
    int i = threadIdx.x;
    float xr = row[KVL + 2 * i], xi = row[KVL + 2 * i + 1];
    float c = fc[s * (DR / 2) + i], sn = fs[s * (DR / 2) + i];
    kpe[(long)s * DR + 2 * i]     = f2bf(xr * c - xi * sn);
    kpe[(long)s * DR + 2 * i + 1] = f2bf(xr * sn + xi * c);
  }
}

// src [R][C] bf16 -> dst [C][R] bf16
__global__ void k_transpose_bf(const u16* __restrict__ src, u16* __restrict__ dst,
                               int R, int C) {
  long i = blockIdx.x * (long)blockDim.x + threadIdx.x;
  long total = (long)R * C;
  if (i >= total) return;
  int  c = (int)(i % C);
  long r = i / C;
  dst[(long)c * R + r] = src[i];
}

// ---------------- generic NT WMMA GEMM (TDM-staged A) ----------------
// C[M][N] = A[M][K] * BT[N][K]^T, bf16 in, f32 accumulate.
// 256 threads = 8 waves; block tile 128(M) x 64(N); wave tile 32x32 (2x2 WMMA).
// A block tile (128 x 64 k-chunk) is DMA'd into LDS by the Tensor Data Mover,
// double-buffered, synchronized with s_wait_tensorcnt + workgroup barrier.
#define KCHUNK 64
template<bool OUT_BF16>
__global__ __launch_bounds__(256)
void gemm_wmma_nt(const u16* __restrict__ A, const u16* __restrict__ BT,
                  void* __restrict__ Cv, int M, int N, int K, int ldc,
                  long aBatch, long bBatch, long cBatch) {
  __shared__ u16 sA[2][128][KCHUNK];   // 2 x 16 KB

  const int lane = threadIdx.x & 31;
  const int wid  = threadIdx.x >> 5;
  const int ln   = lane & 15;
  const int hi   = lane >> 4;        // lane half
  const int akb  = hi ? 16 : 0;      // A K-base, contiguous k order (matches B)
  const int bkb  = hi ? 16 : 0;      // B K-base, contiguous k order (matches A)

  const long zb = blockIdx.z;
  A  += zb * aBatch;
  BT += zb * bBatch;

  const int mBlk = blockIdx.y * 128;
  const int m0   = mBlk + (wid >> 1) * 32;
  const int n0   = blockIdx.x * 64 + (wid & 1) * 32;

  const u16* bR[2];
  bool bOk[2];
  {
    int n1 = n0 + ln, n2 = n0 + 16 + ln;
    bOk[0] = n1 < N; bOk[1] = n2 < N;
    bR[0] = BT + (long)(bOk[0] ? n1 : 0) * K;
    bR[1] = BT + (long)(bOk[1] ? n2 : 0) * K;
  }
  const u16* aBlk = A + (long)mBlk * K;
  const int nStage = K / KCHUNK;

  // stage 0 DMA
  if (wid == 0)
    tdm_load_2d(aBlk, (u32)(size_t)(void*)&sA[0][0][0], 128, KCHUNK, K, M - mBlk);

  v8f acc[2][2] = {};
  for (int st = 0; st < nStage; st++) {
    if (wid == 0) {
      if (st + 1 < nStage) {
        tdm_load_2d(aBlk + (st + 1) * KCHUNK,
                    (u32)(size_t)(void*)&sA[(st + 1) & 1][0][0],
                    128, KCHUNK, K, M - mBlk);
        __builtin_amdgcn_s_wait_tensorcnt(1);   // current stage landed
      } else {
        __builtin_amdgcn_s_wait_tensorcnt(0);
      }
    }
    __syncthreads();

    const u16* aL[2];
    aL[0] = &sA[st & 1][(wid >> 1) * 32 + ln][0];
    aL[1] = &sA[st & 1][(wid >> 1) * 32 + 16 + ln][0];

#pragma unroll
    for (int kk = 0; kk < KCHUNK; kk += 32) {
      const int k0 = st * KCHUNK + kk;
      FragAB a[2], b[2];
#pragma unroll
      for (int i = 0; i < 2; i++) {
#pragma unroll
        for (int v = 0; v < 4; v++) {
          a[i].u[v]     = *(const u32*)(aL[i] + kk + akb + 2 * v);
          a[i].u[v + 4] = *(const u32*)(aL[i] + kk + 8 + akb + 2 * v);
        }
      }
#pragma unroll
      for (int j = 0; j < 2; j++) {
        __builtin_prefetch(bR[j] + k0 + KCHUNK, 0, 1);   // next-stage B stream
#pragma unroll
        for (int v = 0; v < 8; v++) {
          u32 t = *(const u32*)(bR[j] + k0 + bkb + 2 * v);
          b[j].u[v] = bOk[j] ? t : 0u;
        }
      }
#pragma unroll
      for (int i = 0; i < 2; i++)
#pragma unroll
        for (int j = 0; j < 2; j++)
          acc[i][j] = wmma_bf16(a[i], b[j], acc[i][j]);
    }
    __syncthreads();
  }

  if (OUT_BF16) {
    u16* C = (u16*)Cv + zb * cBatch;
#pragma unroll
    for (int i = 0; i < 2; i++)
#pragma unroll
      for (int j = 0; j < 2; j++)
#pragma unroll
        for (int r = 0; r < 8; r++) {
          int m = m0 + i * 16 + r + hi * 8;
          int n = n0 + j * 16 + ln;
          if (m < M && n < N) C[(long)m * ldc + n] = f2bf(acc[i][j][r]);
        }
  } else {
    float* C = (float*)Cv + zb * cBatch;
#pragma unroll
    for (int i = 0; i < 2; i++)
#pragma unroll
      for (int j = 0; j < 2; j++)
#pragma unroll
        for (int r = 0; r < 8; r++) {
          int m = m0 + i * 16 + r + hi * 8;
          int n = n0 + j * 16 + ln;
          if (m < M && n < N) C[(long)m * ldc + n] = acc[i][j][r];
        }
  }
}

// ---------------- fused flash-style MLA attention ----------------
// grid (SEQ/32, NH), 256 threads = 8 waves.
// wave w: mh = w&1 (16-row query half), qd = w>>1 (score 16-key col / O 128-col quarter)
__global__ __launch_bounds__(256)
void mla_attn(const u16* __restrict__ qabs,  // [NH][S][KVL]
              const u16* __restrict__ qpe,   // [NH][S][DR]
              const u16* __restrict__ kvc,   // [S][KVL]
              const u16* __restrict__ kvct,  // [KVL][S]
              const u16* __restrict__ kpe,   // [S][DR]
              u16* __restrict__ omid)        // [NH][S][KVL]
{
  __shared__ float s_sc[32][65];
  __shared__ u16   s_p[32][64];
  __shared__ float s_alpha[32];
  __shared__ float s_linv[32];

  const int lane = threadIdx.x & 31;
  const int wid  = threadIdx.x >> 5;
  const int ln = lane & 15, hi = lane >> 4;
  const int akb = hi ? 16 : 0, bkb = hi ? 16 : 0;
  const int h  = blockIdx.y;
  const int q0 = blockIdx.x * 32;
  const int mh = wid & 1;
  const int qd = wid >> 1;

  const u16* qaR = qabs + ((long)h * SEQ + q0 + mh * 16 + ln) * KVL;
  const u16* qpR = qpe  + ((long)h * SEQ + q0 + mh * 16 + ln) * DR;

  v8f o[8] = {};                       // 16 x 128 O slice
  float m_i = -INFINITY, l_i = 0.f;    // row state (only wave 0, row = lane)

  const int tEnd = q0 + 32;            // causal bound (exclusive)
  for (int t0 = 0; t0 < tEnd; t0 += 64) {
    // -------- scores: 16x16 tile over K = 512 (latent) + 64 (rope) --------
    v8f sacc = {};
    const u16* kcR = kvc + (long)(t0 + qd * 16 + ln) * KVL;
#pragma unroll 4
    for (int k0 = 0; k0 < KVL; k0 += 32) {
      FragAB a, b;
#pragma unroll
      for (int v = 0; v < 4; v++) {
        a.u[v]     = *(const u32*)(qaR + k0 + akb + 2 * v);
        a.u[v + 4] = *(const u32*)(qaR + k0 + 8 + akb + 2 * v);
      }
#pragma unroll
      for (int v = 0; v < 8; v++) b.u[v] = *(const u32*)(kcR + k0 + bkb + 2 * v);
      sacc = wmma_bf16(a, b, sacc);
    }
    const u16* kpR = kpe + (long)(t0 + qd * 16 + ln) * DR;
#pragma unroll
    for (int k0 = 0; k0 < DR; k0 += 32) {
      FragAB a, b;
#pragma unroll
      for (int v = 0; v < 4; v++) {
        a.u[v]     = *(const u32*)(qpR + k0 + akb + 2 * v);
        a.u[v + 4] = *(const u32*)(qpR + k0 + 8 + akb + 2 * v);
      }
#pragma unroll
      for (int v = 0; v < 8; v++) b.u[v] = *(const u32*)(kpR + k0 + bkb + 2 * v);
      sacc = wmma_bf16(a, b, sacc);
    }
#pragma unroll
    for (int r = 0; r < 8; r++) {
      int mloc = mh * 16 + r + hi * 8;
      int tg   = t0 + qd * 16 + ln;
      float v  = sacc[r] * SCALE_F;
      if (tg > q0 + mloc) v = -INFINITY;
      s_sc[mloc][qd * 16 + ln] = v;
    }
    __syncthreads();

    // -------- online softmax (wave 0, lane = row) --------
    if (wid == 0) {
      int row = lane;
      float mx = m_i;
#pragma unroll 8
      for (int j = 0; j < 64; j++) mx = fmaxf(mx, s_sc[row][j]);
      float alpha = __expf(m_i - mx);
      float sum = 0.f;
#pragma unroll 8
      for (int j = 0; j < 64; j++) {
        float p = __expf(s_sc[row][j] - mx);
        sum += p;
        s_p[row][j] = f2bf(p);
      }
      m_i = mx;
      l_i = l_i * alpha + sum;
      s_alpha[row] = alpha;
    }
    __syncthreads();

    // -------- rescale O, then O += P @ V (V = kv_c slice) --------
#pragma unroll
    for (int r = 0; r < 8; r++) {
      float a = s_alpha[mh * 16 + r + hi * 8];
#pragma unroll
      for (int nj = 0; nj < 8; nj++) o[nj][r] *= a;
    }
    const u16* pRow = &s_p[mh * 16 + ln][0];
#pragma unroll
    for (int kt = 0; kt < 64; kt += 32) {
      FragAB pa;
#pragma unroll
      for (int v = 0; v < 4; v++) {
        pa.u[v]     = *(const u32*)(pRow + kt + akb + 2 * v);
        pa.u[v + 4] = *(const u32*)(pRow + kt + 8 + akb + 2 * v);
      }
#pragma unroll
      for (int nj = 0; nj < 8; nj++) {
        FragAB vb;
        const u16* vRow = kvct + (long)(qd * 128 + nj * 16 + ln) * SEQ + t0;
#pragma unroll
        for (int v = 0; v < 8; v++)
          vb.u[v] = *(const u32*)(vRow + kt + bkb + 2 * v);
        o[nj] = wmma_bf16(pa, vb, o[nj]);
      }
    }
    __syncthreads();
  }

  if (wid == 0) s_linv[lane] = 1.f / l_i;
  __syncthreads();

  u16* oR = omid + ((long)h * SEQ + q0) * KVL;
#pragma unroll
  for (int r = 0; r < 8; r++) {
    int mloc = mh * 16 + r + hi * 8;
    float inv = s_linv[mloc];
#pragma unroll
    for (int nj = 0; nj < 8; nj++)
      oR[(long)mloc * KVL + qd * 128 + nj * 16 + ln] = f2bf(o[nj][r] * inv);
  }
}

// ---------------- host orchestration ----------------
extern "C" void kernel_launch(void* const* d_in, const int* in_sizes, int n_in,
                              void* d_out, int out_size, void* d_ws, size_t ws_size,
                              hipStream_t stream) {
  (void)in_sizes; (void)n_in; (void)out_size; (void)ws_size;
  const float* x    = (const float*)d_in[0];
  const float* wq   = (const float*)d_in[1];
  const float* wkva = (const float*)d_in[2];
  const float* wkvb = (const float*)d_in[3];
  const float* kvnw = (const float*)d_in[4];
  const float* wo   = (const float*)d_in[5];
  const float* fcos = (const float*)d_in[6];
  const float* fsin = (const float*)d_in[7];
  // d_in[8] = start_pos (always 0 in this problem)

  char* ws = (char*)d_ws;
  size_t off = 0;
  auto take = [&](size_t bytes) -> char* {
    char* p = ws + off;
    off += (bytes + 255) & ~(size_t)255;
    return p;
  };
  u16*   x_bf    = (u16*)take((size_t)SEQ * DIMX * 2);
  u16*   wq_t    = (u16*)take((size_t)NQ * DIMX * 2);
  u16*   wkva_t  = (u16*)take((size_t)KVA * DIMX * 2);
  u16*   wbk_t   = (u16*)take((size_t)NH * KVL * DN * 2);
  u16*   wbv     = (u16*)take((size_t)NH * DV * KVL * 2);
  u16*   wo_t    = (u16*)take((size_t)DIMX * ODIM * 2);
  u16*   q_bf    = (u16*)take((size_t)SEQ * NQ * 2);
  float* kv_f32  = (float*)take((size_t)SEQ * KVA * 4);
  u16*   qn      = (u16*)take((size_t)NH * SEQ * DN * 2);
  u16*   qp      = (u16*)take((size_t)NH * SEQ * DR * 2);
  u16*   kvc     = (u16*)take((size_t)SEQ * KVL * 2);
  u16*   kvct    = (u16*)take((size_t)KVL * SEQ * 2);
  u16*   kpe     = (u16*)take((size_t)SEQ * DR * 2);
  u16*   qabs    = (u16*)take((size_t)NH * SEQ * KVL * 2);
  u16*   omid    = (u16*)take((size_t)NH * SEQ * KVL * 2);
  u16*   oheads  = (u16*)take((size_t)SEQ * ODIM * 2);

  auto blks = [](long n) { return (unsigned)((n + 255) / 256); };

  // precision staging (f32 -> bf16; weights transposed to [N][K])
  k_f32_to_bf16<<<blks((long)SEQ * DIMX), 256, 0, stream>>>(x, x_bf, (long)SEQ * DIMX);
  k_convert_t<<<blks((long)DIMX * NQ), 256, 0, stream>>>(wq, wq_t, DIMX, NQ);
  k_convert_t<<<blks((long)DIMX * KVA), 256, 0, stream>>>(wkva, wkva_t, DIMX, KVA);
  k_split_wb<<<blks((long)NH * (DN + DV) * KVL), 256, 0, stream>>>(wkvb, wbk_t, wbv);
  k_convert_t<<<blks((long)ODIM * DIMX), 256, 0, stream>>>(wo, wo_t, ODIM, DIMX);

  // q = x @ wq  (bf16 out)
  gemm_wmma_nt<true><<<dim3(NQ / 64, SEQ / 128, 1), 256, 0, stream>>>(
      x_bf, wq_t, q_bf, SEQ, NQ, DIMX, NQ, 0, 0, 0);
  // kv_full = x @ wkv_a  (f32 out, feeds rms-norm)
  gemm_wmma_nt<false><<<dim3((KVA + 63) / 64, SEQ / 128, 1), 256, 0, stream>>>(
      x_bf, wkva_t, kv_f32, SEQ, KVA, DIMX, KVA, 0, 0, 0);

  k_rope_q<<<blks((long)SEQ * NH), 256, 0, stream>>>(q_bf, fcos, fsin, qn, qp);
  k_kv_post<<<SEQ, 256, 0, stream>>>(kv_f32, kvnw, fcos, fsin, kvc, kpe);
  k_transpose_bf<<<blks((long)SEQ * KVL), 256, 0, stream>>>(kvc, kvct, SEQ, KVL);

  // q_abs[h] = qn[h] @ wb_k[h]   (batched over heads)
  gemm_wmma_nt<true><<<dim3(KVL / 64, SEQ / 128, NH), 256, 0, stream>>>(
      qn, wbk_t, qabs, SEQ, KVL, DN, KVL,
      (long)SEQ * DN, (long)KVL * DN, (long)SEQ * KVL);

  // fused attention: scores + online softmax + P@kv_c
  mla_attn<<<dim3(SEQ / 32, NH), 256, 0, stream>>>(qabs, qp, kvc, kvct, kpe, omid);

  // o_heads[:, h*128:(h+1)*128] = omid[h] @ wb_v[h]^T  (batched over heads)
  gemm_wmma_nt<true><<<dim3(DV / 64, SEQ / 128, NH), 256, 0, stream>>>(
      omid, wbv, oheads, SEQ, DV, KVL, ODIM,
      (long)SEQ * KVL, (long)DV * KVL, (long)DV);

  // out = o_heads @ wo  (f32 out)
  gemm_wmma_nt<false><<<dim3(DIMX / 64, SEQ / 128, 1), 256, 0, stream>>>(
      oheads, wo_t, d_out, SEQ, DIMX, ODIM, DIMX, 0, 0, 0);
}
